// InfoNCELoss_22376779612598
// MI455X (gfx1250) — matrix operand order repaired and checked by reading, not compiled, not run on hardware
//
#include <hip/hip_runtime.h>
#include <hip/hip_bf16.h>

typedef __attribute__((ext_vector_type(16))) _Float16 v16h;
typedef __attribute__((ext_vector_type(8)))  float    v8f;
typedef int v4i_ __attribute__((vector_size(16)));

#define TEMP_INV  (1.0f / 0.07f)
#define TILE      128
#define KC        64   // K-chunk depth per LDS stage
#define STR       72   // LDS row stride in halves (144B, 16B-aligned, padded)

#if defined(__gfx1250__) && __has_builtin(__builtin_amdgcn_global_load_async_to_lds_b128)
#define HAS_ASYNC 1
#else
#define HAS_ASYNC 0
#endif

#if HAS_ASYNC
typedef __attribute__((address_space(1))) v4i_* gvec_ptr;
typedef __attribute__((address_space(3))) v4i_* lvec_ptr;
#  if __has_builtin(__builtin_amdgcn_s_wait_asynccnt)
#    define WAIT_ASYNC(n) __builtin_amdgcn_s_wait_asynccnt(n)
#  else
#    define WAIT_ASYNC(n) asm volatile("s_wait_asynccnt " #n ::: "memory")
#  endif
#else
#  define WAIT_ASYNC(n) ((void)0)
#endif

// ---------------- row L2-normalize fp32 -> f16 ----------------
__global__ __launch_bounds__(256) void norm_kernel(const float* __restrict__ X,
                                                   _Float16* __restrict__ Y, int D) {
    int row = blockIdx.x;
    const float* x = X + (size_t)row * D;
    float ss = 0.f;
    for (int k = threadIdx.x; k < D; k += 256) { float v = x[k]; ss += v * v; }
    #pragma unroll
    for (int off = 16; off; off >>= 1) ss += __shfl_xor(ss, off, 32);
    __shared__ float red[8];
    if ((threadIdx.x & 31) == 0) red[threadIdx.x >> 5] = ss;
    __syncthreads();
    float tot = 0.f;
    #pragma unroll
    for (int w = 0; w < 8; w++) tot += red[w];
    float inv = 1.0f / fmaxf(sqrtf(tot), 1e-12f);
    _Float16* y = Y + (size_t)row * D;
    for (int k = threadIdx.x; k < D; k += 256) y[k] = (_Float16)(x[k] * inv);
}

// ---------------- zero-init accumulators ----------------
__global__ __launch_bounds__(256) void zero_kernel(float* __restrict__ p, int n) {
    int i = blockIdx.x * 256 + threadIdx.x;
    if (i < n) p[i] = 0.f;
}

// ---------------- async (or fallback) global -> LDS tile copy ----------------
// Each thread moves 64B: row = tid>>1, khalf base = (tid&1)*32, 4 x 16B chunks.
__device__ inline void stage_tile(const _Float16* __restrict__ g, int D, int gRowBase,
                                  int kt, _Float16* lbuf, int ldRow, int ldK) {
    const char* gp = (const char*)(g + (size_t)(gRowBase + ldRow) * D + kt + ldK);
    _Float16*   lp = lbuf + ldRow * STR + ldK;
#if HAS_ASYNC
    #pragma unroll
    for (int c = 0; c < 4; ++c)
        __builtin_amdgcn_global_load_async_to_lds_b128(
            (gvec_ptr)(gp + c * 16),
            (lvec_ptr)(lp + c * 8),
            0, 0);
#else
    #pragma unroll
    for (int c = 0; c < 4; ++c)
        *(float4*)(lp + c * 8) = *(const float4*)(gp + c * 16);
#endif
}

// ---------------- fragment loads from LDS (wave32 WMMA layouts) ----------------
// A 16x32 f16: lanes 0-15 row m=lane, K=0..15 contiguous;
//              lanes 16-31 same rows, K=16..31.
__device__ inline v16h load_fragA(const _Float16* lds, int row0, int ks, int lane) {
    int h   = lane >> 4;
    int idx = lane & 15;
    const _Float16* p = lds + (row0 + idx) * STR + ks;
    v16h f;
    *(((float4*)&f) + 0) = *(const float4*)(p + h * 16);
    *(((float4*)&f) + 1) = *(const float4*)(p + h * 16 + 8);
    return f;
}
// B 32x16 f16 (col n striped across lanes): lanes 0-15 hold K=0..15 of col n,
// lanes 16-31 hold K=16..31; contiguous K, 2 halves per VGPR.
__device__ inline v16h load_fragB(const _Float16* lds, int col0, int ks, int lane) {
    int h   = lane >> 4;
    int idx = lane & 15;
    const _Float16* p = lds + (col0 + idx) * STR + ks + h * 16;
    v16h f;
    *(((float4*)&f) + 0) = *(const float4*)(p + 0);
    *(((float4*)&f) + 1) = *(const float4*)(p + 8);
    return f;
}

// ---------------- fused WMMA GEMM + exp + row-sum ----------------
__global__ __launch_bounds__(256) void sim_kernel(const _Float16* __restrict__ A,
                                                  const _Float16* __restrict__ B,
                                                  const long long* __restrict__ la,
                                                  const long long* __restrict__ lb,
                                                  float* __restrict__ pos,
                                                  float* __restrict__ neg,
                                                  int D) {
    __shared__ _Float16 sA[2][TILE * STR];
    __shared__ _Float16 sB[2][TILE * STR];

    const int tid  = threadIdx.x;
    const int lane = tid & 31;
    const int wave = tid >> 5;
    const int wr   = wave & 3;   // 4 row-groups of 32 rows
    const int wc   = wave >> 2;  // 2 col-groups of 64 cols

    const int rowBase = blockIdx.y * TILE;
    const int colBase = blockIdx.x * TILE;

    v8f acc[2][4] = {};

    const int ldRow = tid >> 1;          // 0..127
    const int ldK   = (tid & 1) * 32;    // halves 0 or 32

    const int nIter = D / KC;            // 16

    // prologue: stage chunk 0 into buffer 0
    stage_tile(A, D, rowBase, 0, sA[0], ldRow, ldK);
    stage_tile(B, D, colBase, 0, sB[0], ldRow, ldK);

    for (int it = 0; it < nIter; ++it) {
        if (it + 1 < nIter) {
            // prefetch next chunk into the other buffer (overlaps with compute below)
            stage_tile(A, D, rowBase, (it + 1) * KC, sA[(it + 1) & 1], ldRow, ldK);
            stage_tile(B, D, colBase, (it + 1) * KC, sB[(it + 1) & 1], ldRow, ldK);
            WAIT_ASYNC(8);   // this wave's chunk-it loads (8 oldest) have landed
        } else {
            WAIT_ASYNC(0);
        }
        __syncthreads();     // every wave's chunk-it data resident in LDS

        const _Float16* bufA = sA[it & 1];
        const _Float16* bufB = sB[it & 1];
        #pragma unroll
        for (int ks = 0; ks < KC; ks += 32) {
            v16h af[2], bf[4];
            #pragma unroll
            for (int i = 0; i < 2; i++) af[i] = load_fragA(bufA, wr * 32 + i * 16, ks, lane);
            #pragma unroll
            for (int j = 0; j < 4; j++) bf[j] = load_fragB(bufB, wc * 64 + j * 16, ks, lane);
            #pragma unroll
            for (int i = 0; i < 2; i++)
                #pragma unroll
                for (int j = 0; j < 4; j++)
                    acc[i][j] = __builtin_amdgcn_wmma_f32_16x16x32_f16(
                        false, af[i], false, bf[j], (short)0, acc[i][j], false, false);
        }
        __syncthreads();     // all waves done reading buf[it&1] before it is refilled
    }

    // epilogue: scale, clip, exp, masked row sums, atomics
    const int h    = lane >> 4;
    const int nIdx = lane & 15;

    long long lbj[4];
    #pragma unroll
    for (int j = 0; j < 4; j++)
        lbj[j] = lb[colBase + wc * 64 + j * 16 + nIdx];

    #pragma unroll
    for (int i = 0; i < 2; i++) {
        #pragma unroll
        for (int v = 0; v < 8; v++) {
            const int row = rowBase + wr * 32 + i * 16 + h * 8 + v;
            const long long myla = la[row];
            float nsum = 0.f, psum = 0.f;
            #pragma unroll
            for (int j = 0; j < 4; j++) {
                float s = acc[i][j][v] * TEMP_INV;
                s = fminf(fmaxf(s, -50.f), 50.f);
                float e = __expf(s);
                nsum += e;
                if (lbj[j] == myla) psum += e;
            }
            #pragma unroll
            for (int off = 1; off < 16; off <<= 1) {
                nsum += __shfl_xor(nsum, off, 32);
                psum += __shfl_xor(psum, off, 32);
            }
            if (nIdx == 0) {
                atomicAdd(&neg[row], nsum);
                atomicAdd(&pos[row], psum);
            }
        }
    }
}

// ---------------- final loss reduction ----------------
__global__ __launch_bounds__(1024) void loss_kernel(const float* __restrict__ pos,
                                                    const float* __restrict__ neg,
                                                    float* __restrict__ out, int N) {
    float acc = 0.f;
    for (int i = threadIdx.x; i < N; i += 1024) {
        float p = fmaxf(pos[i], 1e-8f);
        acc += logf(neg[i]) - logf(p);   // -log(p/n)
    }
    #pragma unroll
    for (int off = 16; off; off >>= 1) acc += __shfl_xor(acc, off, 32);
    __shared__ float red[32];
    if ((threadIdx.x & 31) == 0) red[threadIdx.x >> 5] = acc;
    __syncthreads();
    if (threadIdx.x < 32) {
        float t = red[threadIdx.x];
        #pragma unroll
        for (int off = 16; off; off >>= 1) t += __shfl_xor(t, off, 32);
        if (threadIdx.x == 0) out[0] = t / (float)N;
    }
}

extern "C" void kernel_launch(void* const* d_in, const int* in_sizes, int n_in,
                              void* d_out, int out_size, void* d_ws, size_t ws_size,
                              hipStream_t stream) {
    const float*     fa = (const float*)d_in[0];
    const float*     fb = (const float*)d_in[1];
    const long long* la = (const long long*)d_in[2];
    const long long* lb = (const long long*)d_in[3];

    const int N = in_sizes[2];          // 8192
    const int M = in_sizes[3];          // 8192
    const int D = in_sizes[0] / N;      // 1024

    char* ws = (char*)d_ws;
    _Float16* Ah = (_Float16*)ws;
    _Float16* Bh = (_Float16*)(ws + (size_t)N * D * sizeof(_Float16));
    float*    pos = (float*)(ws + (size_t)(N + M) * D * sizeof(_Float16));
    float*    neg = pos + N;

    norm_kernel<<<N, 256, 0, stream>>>(fa, Ah, D);
    norm_kernel<<<M, 256, 0, stream>>>(fb, Bh, D);
    zero_kernel<<<(2 * N + 255) / 256, 256, 0, stream>>>(pos, 2 * N);

    dim3 grid(M / TILE, N / TILE);
    sim_kernel<<<grid, 256, 0, stream>>>(Ah, Bh, la, lb, pos, neg, D);

    loss_kernel<<<1, 1024, 0, stream>>>(pos, neg, (float*)d_out, N);
}
